// DualMemoryElmanCell_39960375722199
// MI455X (gfx1250) — compile-verified, matrix operand order checked by reading, not run on hardware
//
#include <hip/hip_runtime.h>
#include <hip/hip_bf16.h>

// Problem constants (from reference): B=32, T=1024, D=1024, N=64
#define BB 32
#define TT 1024
#define DD 1024
#define NN 64
#define ATT_SCALE 0.03125f  // 1/sqrt(1024)

typedef __attribute__((ext_vector_type(16))) __bf16 v16bf;
typedef __attribute__((ext_vector_type(8)))  __bf16 v8bf;
typedef __attribute__((ext_vector_type(8)))  float  v8f;

// ---------------------------------------------------------------------------
// Fragment loaders for v_wmma_f32_16x16x32_bf16 (wave32)
//
// A (16x32 bf16, row-major, stride DD): lane l holds row (l&15); element e of
// the 16 bf16 holds K = (e>=8?16:0) + (l>>4)*8 + (e&7)  ->  with k0 =
// kt*32 + (l>>4)*8 this is two contiguous 8-element (16B) chunks at k0 and
// k0+16.  B (32x16) taken from row-major W[n,k] uses the identical pattern
// with N as the lane-striped index.
// ---------------------------------------------------------------------------
__device__ __forceinline__ v16bf load_frag_bf(const __bf16* __restrict__ base,
                                              int row, int k0) {
  const v8bf* p0 = reinterpret_cast<const v8bf*>(base + (size_t)row * DD + k0);
  const v8bf* p1 = reinterpret_cast<const v8bf*>(base + (size_t)row * DD + k0 + 16);
  v8bf lo = *p0, hi = *p1;
  v16bf r;
#pragma unroll
  for (int i = 0; i < 8; ++i) { r[i] = lo[i]; r[i + 8] = hi[i]; }
  return r;
}

__device__ __forceinline__ v16bf load_frag_f32(const float* __restrict__ base,
                                               int row, int k0) {
  const float* p = base + (size_t)row * DD + k0;
  v16bf r;
#pragma unroll
  for (int i = 0; i < 8; ++i) {
    r[i]     = (__bf16)p[i];
    r[i + 8] = (__bf16)p[16 + i];
  }
  return r;
}

// ---------------------------------------------------------------------------
// Elementwise helpers
// ---------------------------------------------------------------------------
__global__ void __launch_bounds__(256)
f32_to_bf16_kernel(const float* __restrict__ src, __bf16* __restrict__ dst, int n) {
  int i = blockIdx.x * 256 + threadIdx.x;
  if (i < n) dst[i] = (__bf16)src[i];
}

__global__ void __launch_bounds__(256)
copy_f32_kernel(const float* __restrict__ src, float* __restrict__ dst, int n) {
  int i = blockIdx.x * 256 + threadIdx.x;
  if (i < n) dst[i] = src[i];
}

// ---------------------------------------------------------------------------
// x_proj GEMM: Y[m,e] = sum_d X[m,d] * W[e,d];  M = B*T, N = K = D.
// Block = 256 thr = 8 waves arranged 2 (M) x 4 (N): wg covers 32 rows x 64
// cols.  Grid = (D/64, M/32).  A converted f32->bf16 in registers (x_seq is
// L2-resident fp32); W pre-converted to bf16.
// ---------------------------------------------------------------------------
__global__ void __launch_bounds__(256)
xproj_gemm_kernel(const float* __restrict__ X, const __bf16* __restrict__ W,
                  float* __restrict__ Y) {
  const int tid  = threadIdx.x;
  const int wave = tid >> 5, lane = tid & 31;
  const int lr = lane & 15, lh = lane >> 4;
  const int wm = wave & 1, wn = wave >> 1;
  const int rowbase = blockIdx.y * 32 + wm * 16;
  const int colbase = blockIdx.x * 64 + wn * 16;
  const int arow = rowbase + lr;
  const int brow = colbase + lr;  // row of W == output column

  v8f acc = {};
#pragma unroll 4
  for (int kt = 0; kt < DD / 32; ++kt) {
    const int k0 = kt * 32 + lh * 8;
    __builtin_prefetch(W + (size_t)brow * DD + k0 + 128, 0, 1);
    v16bf a = load_frag_f32(X, arow, k0);
    v16bf b = load_frag_bf(W, brow, k0);
    acc = __builtin_amdgcn_wmma_f32_16x16x32_bf16(false, a, false, b,
                                                  (short)0, acc, false, false);
  }
  const int col = colbase + lr;
#pragma unroll
  for (int r = 0; r < 8; ++r) {
    const int m = rowbase + lh * 8 + r;
    Y[(size_t)m * DD + col] = acc[r];
  }
}

// ---------------------------------------------------------------------------
// Attention kernel: one workgroup per batch b.
//   s[n]   = (h . tape[b,n,:]) * scale ; attn = softmax(s)
//   read_out[b,:] = sum_n attn[n] * tape[b,n,:]          (if read_out != null)
//   attn_out[b,:] = attn                                  (if attn_out != null)
// ---------------------------------------------------------------------------
__global__ void __launch_bounds__(256)
attn_kernel(const float* __restrict__ hw, const float* __restrict__ tape,
            float* __restrict__ attn_out, float* __restrict__ read_out) {
  const int b = blockIdx.x, tid = threadIdx.x;
  __shared__ float part[NN][4];
  __shared__ float attn[NN];
  const float* hwb = hw + (size_t)b * DD;
  const float* tb  = tape + (size_t)b * NN * DD;

  const int n = tid & 63, q = tid >> 6;      // 4 threads per tape row
  {
    const float* tr = tb + (size_t)n * DD + q * 256;
    const float* hq = hwb + q * 256;
    float s = 0.f;
#pragma unroll 4
    for (int d = 0; d < 256; ++d) s = fmaf(tr[d], hq[d], s);
    part[n][q] = s;
  }
  __syncthreads();
  if (tid < NN)
    attn[tid] = (part[tid][0] + part[tid][1] + part[tid][2] + part[tid][3]) * ATT_SCALE;
  __syncthreads();
  if (tid == 0) {
    float m = attn[0];
    for (int i = 1; i < NN; ++i) m = fmaxf(m, attn[i]);
    float sum = 0.f;
    for (int i = 0; i < NN; ++i) { float e = __expf(attn[i] - m); attn[i] = e; sum += e; }
    float inv = 1.f / sum;
    for (int i = 0; i < NN; ++i) attn[i] *= inv;
  }
  __syncthreads();
  if (attn_out && tid < NN) attn_out[b * NN + tid] = attn[tid];
  if (read_out) {
    for (int d = tid; d < DD; d += 256) {
      float acc = 0.f;
#pragma unroll 8
      for (int nn = 0; nn < NN; ++nn) acc = fmaf(attn[nn], tb[(size_t)nn * DD + d], acc);
      read_out[(size_t)b * DD + d] = acc;
    }
  }
}

// ---------------------------------------------------------------------------
// Work-state GEMM + tanh:  h_new = tanh(h_prev @ W_h^T + x_t + read + b_h)
// M = B = 32, N = K = D.  Grid = D/64 wgs, 8 waves: 2 (M) x 4 (N).
// Writes h_new to d_out (h_work_all[b,t,:]), fp32 next-state, bf16 next-state.
// ---------------------------------------------------------------------------
__global__ void __launch_bounds__(256)
work_gemm_kernel(const __bf16* __restrict__ Abf, const __bf16* __restrict__ Wbf,
                 const float* __restrict__ xproj, const float* __restrict__ readbuf,
                 const float* __restrict__ bh, float* __restrict__ hall,
                 float* __restrict__ hf_next, __bf16* __restrict__ hb_next, int t) {
  const int tid  = threadIdx.x;
  const int wave = tid >> 5, lane = tid & 31;
  const int lr = lane & 15, lh = lane >> 4;
  const int wm = wave & 1, wn = wave >> 1;
  const int rowbase = wm * 16;
  const int colbase = blockIdx.x * 64 + wn * 16;
  const int arow = rowbase + lr;
  const int brow = colbase + lr;

  v8f acc = {};
#pragma unroll 4
  for (int kt = 0; kt < DD / 32; ++kt) {
    const int k0 = kt * 32 + lh * 8;
    __builtin_prefetch(Wbf + (size_t)brow * DD + k0 + 128, 0, 1);
    v16bf a = load_frag_bf(Abf, arow, k0);
    v16bf b = load_frag_bf(Wbf, brow, k0);
    acc = __builtin_amdgcn_wmma_f32_16x16x32_bf16(false, a, false, b,
                                                  (short)0, acc, false, false);
  }
  const int col = colbase + lr;
#pragma unroll
  for (int r = 0; r < 8; ++r) {
    const int bb = rowbase + lh * 8 + r;  // batch index
    float v = acc[r] + xproj[((size_t)bb * TT + t) * DD + col]
                     + readbuf[(size_t)bb * DD + col] + bh[col];
    float h = tanhf(v);
    hall[((size_t)bb * TT + t) * DD + col] = h;
    hf_next[(size_t)bb * DD + col] = h;
    hb_next[(size_t)bb * DD + col] = (__bf16)h;
  }
}

// ---------------------------------------------------------------------------
// Write-value GEMM + tape update.
//   wv = h_new @ W_write^T           (WMMA, staged to LDS)
//   tape[b,n,col] += w[b,n] * (wv[b,col] - tape[b,n,col])
// Each wg owns a disjoint 64-column slab of the tape.  After the GEMM, the
// 32x64 wv tile sits in LDS and all 256 threads re-partition the RMW over the
// 2048 (b,n) tape rows with float4 (b128) accesses over 64 contiguous cols.
// ---------------------------------------------------------------------------
__global__ void __launch_bounds__(256)
write_update_kernel(const __bf16* __restrict__ Abf, const __bf16* __restrict__ Wbf,
                    const float* __restrict__ wattn, float* __restrict__ tape) {
  const int tid  = threadIdx.x;
  const int wave = tid >> 5, lane = tid & 31;
  const int lr = lane & 15, lh = lane >> 4;
  const int wm = wave & 1, wn = wave >> 1;
  const int rowbase = wm * 16;
  const int colbase = blockIdx.x * 64 + wn * 16;
  const int arow = rowbase + lr;
  const int brow = colbase + lr;

  __shared__ float wv_s[BB][64];   // 8 KB write_value tile for this column slab

  v8f acc = {};
#pragma unroll 4
  for (int kt = 0; kt < DD / 32; ++kt) {
    const int k0 = kt * 32 + lh * 8;
    __builtin_prefetch(Wbf + (size_t)brow * DD + k0 + 128, 0, 1);
    v16bf a = load_frag_bf(Abf, arow, k0);
    v16bf b = load_frag_bf(Wbf, brow, k0);
    acc = __builtin_amdgcn_wmma_f32_16x16x32_bf16(false, a, false, b,
                                                  (short)0, acc, false, false);
  }
  const int lcol = wn * 16 + lr;   // column within this wg's 64-col slab
#pragma unroll
  for (int r = 0; r < 8; ++r) {
    const int bb = rowbase + lh * 8 + r;
    wv_s[bb][lcol] = acc[r];
  }
  __syncthreads();

  // Coalesced tape blend: 2048 (b,n) rows / 256 threads = 8 rows per thread,
  // each row 64 contiguous floats -> 16 float4 RMWs.
  const int colg0 = blockIdx.x * 64;
  for (int rowi = tid; rowi < BB * NN; rowi += 256) {
    const int bb = rowi >> 6, nn = rowi & 63;
    const float w = wattn[rowi];              // wattn is [B*N] contiguous
    float* tp = tape + ((size_t)bb * NN + nn) * DD + colg0;
    const float* wvp = wv_s[bb];
#pragma unroll
    for (int c = 0; c < 64; c += 4) {
      float4 old = *reinterpret_cast<float4*>(tp + c);
      old.x += w * (wvp[c + 0] - old.x);
      old.y += w * (wvp[c + 1] - old.y);
      old.z += w * (wvp[c + 2] - old.z);
      old.w += w * (wvp[c + 3] - old.w);
      *reinterpret_cast<float4*>(tp + c) = old;
    }
  }
}

// ---------------------------------------------------------------------------
// Host-side orchestration
// ---------------------------------------------------------------------------
static constexpr size_t align256(size_t x) { return (x + 255) & ~(size_t)255; }

extern "C" void kernel_launch(void* const* d_in, const int* in_sizes, int n_in,
                              void* d_out, int out_size, void* d_ws, size_t ws_size,
                              hipStream_t stream) {
  (void)in_sizes; (void)n_in; (void)out_size; (void)ws_size;

  // Inputs per setup_inputs() order
  const float* x_seq   = (const float*)d_in[0];  // [B,T,D]
  const float* h_tape0 = (const float*)d_in[1];  // [B,N,D]
  const float* h_work0 = (const float*)d_in[2];  // [B,D]
  const float* W_h     = (const float*)d_in[3];  // [D,D]
  const float* W_x     = (const float*)d_in[4];  // [D,D]
  const float* b_h     = (const float*)d_in[5];  // [D]
  const float* W_write = (const float*)d_in[6];  // [D,D]

  float* out = (float*)d_out;
  float* out_hall  = out;                                        // [B,T,D]
  float* out_tape  = out + (size_t)BB * TT * DD;                 // [B,N,D]
  float* out_hlast = out_tape + (size_t)BB * NN * DD;            // [B,D]

  // Workspace layout
  char* ws = (char*)d_ws;
  constexpr size_t SZ_XPROJ = (size_t)BB * TT * DD * 4;
  constexpr size_t SZ_W     = (size_t)DD * DD * 2;
  constexpr size_t SZ_TAPE  = (size_t)BB * NN * DD * 4;
  constexpr size_t SZ_HF    = (size_t)BB * DD * 4;
  constexpr size_t SZ_HB    = (size_t)BB * DD * 2;
  constexpr size_t SZ_READ  = (size_t)BB * DD * 4;
  constexpr size_t SZ_WATTN = (size_t)BB * NN * 4;

  constexpr size_t OFF_XPROJ = 0;
  constexpr size_t OFF_WH    = OFF_XPROJ + align256(SZ_XPROJ);
  constexpr size_t OFF_WX    = OFF_WH    + align256(SZ_W);
  constexpr size_t OFF_WW    = OFF_WX    + align256(SZ_W);
  constexpr size_t OFF_TAPE  = OFF_WW    + align256(SZ_W);
  constexpr size_t OFF_HF0   = OFF_TAPE  + align256(SZ_TAPE);
  constexpr size_t OFF_HF1   = OFF_HF0   + align256(SZ_HF);
  constexpr size_t OFF_HB0   = OFF_HF1   + align256(SZ_HF);
  constexpr size_t OFF_HB1   = OFF_HB0   + align256(SZ_HB);
  constexpr size_t OFF_READ  = OFF_HB1   + align256(SZ_HB);
  constexpr size_t OFF_WATTN = OFF_READ  + align256(SZ_READ);

  float*  xproj  = (float*)(ws + OFF_XPROJ);
  __bf16* Wh_bf  = (__bf16*)(ws + OFF_WH);
  __bf16* Wx_bf  = (__bf16*)(ws + OFF_WX);
  __bf16* Ww_bf  = (__bf16*)(ws + OFF_WW);
  float*  tape   = (float*)(ws + OFF_TAPE);
  float*  hf0    = (float*)(ws + OFF_HF0);
  float*  hf1    = (float*)(ws + OFF_HF1);
  __bf16* hb0    = (__bf16*)(ws + OFF_HB0);
  __bf16* hb1    = (__bf16*)(ws + OFF_HB1);
  float*  readbf = (float*)(ws + OFF_READ);
  float*  wattn  = (float*)(ws + OFF_WATTN);

  // --- Prologue: convert weights to bf16, stage mutable state ---------------
  const int nW = DD * DD;
  f32_to_bf16_kernel<<<(nW + 255) / 256, 256, 0, stream>>>(W_h,     Wh_bf, nW);
  f32_to_bf16_kernel<<<(nW + 255) / 256, 256, 0, stream>>>(W_x,     Wx_bf, nW);
  f32_to_bf16_kernel<<<(nW + 255) / 256, 256, 0, stream>>>(W_write, Ww_bf, nW);
  const int nTape = BB * NN * DD;
  copy_f32_kernel<<<(nTape + 255) / 256, 256, 0, stream>>>(h_tape0, tape, nTape);
  const int nH = BB * DD;
  copy_f32_kernel<<<(nH + 255) / 256, 256, 0, stream>>>(h_work0, hf0, nH);
  f32_to_bf16_kernel<<<(nH + 255) / 256, 256, 0, stream>>>(h_work0, hb0, nH);

  // --- Phase 1: x_proj = x_seq @ W_x^T (WMMA bf16, fp32 accumulate) ---------
  {
    dim3 grid(DD / 64, (BB * TT) / 32);
    xproj_gemm_kernel<<<grid, 256, 0, stream>>>(x_seq, Wx_bf, xproj);
  }

  // --- Phase 2: sequential scan (4 small kernels per step, double-buffered) -
  for (int t = 0; t < TT; ++t) {
    float*  hf_prev = (t & 1) ? hf1 : hf0;
    float*  hf_next = (t & 1) ? hf0 : hf1;
    __bf16* hb_prev = (t & 1) ? hb1 : hb0;
    __bf16* hb_next = (t & 1) ? hb0 : hb1;

    // read attention + read vector (uses h_prev, tape pre-update)
    attn_kernel<<<BB, 256, 0, stream>>>(hf_prev, tape, nullptr, readbf);
    // h_new = tanh(h_prev@W_h^T + x_t + read + b_h)
    work_gemm_kernel<<<DD / 64, 256, 0, stream>>>(hb_prev, Wh_bf, xproj, readbf,
                                                  b_h, out_hall, hf_next, hb_next, t);
    // write attention (uses h_new, tape pre-update)
    attn_kernel<<<BB, 256, 0, stream>>>(hf_next, tape, wattn, nullptr);
    // write_value = h_new@W_write^T ; blend into tape
    write_update_kernel<<<DD / 64, 256, 0, stream>>>(hb_next, Ww_bf, wattn, tape);
  }

  // --- Epilogue: emit final tape and final work state -----------------------
  copy_f32_kernel<<<(nTape + 255) / 256, 256, 0, stream>>>(tape, out_tape, nTape);
  // T=1024 is even -> final state landed back in buffer 0
  copy_f32_kernel<<<(nH + 255) / 256, 256, 0, stream>>>(hf0, out_hlast, nH);
}
